// SwinBlock_9637906612549
// MI455X (gfx1250) — matrix operand order repaired and checked by reading, not compiled, not run on hardware
//
#include <hip/hip_runtime.h>
#include <math.h>

typedef _Float16 f16;
typedef __attribute__((ext_vector_type(16))) _Float16 v16h;
typedef __attribute__((ext_vector_type(4)))  _Float16 v4h;
typedef __attribute__((ext_vector_type(8)))  float    v8f;

// Problem constants (match reference)
#define BATCH 64
#define SDIM  56
#define DMODEL 128
#define NHEAD 4
#define WSZ   7
#define WS    49      // tokens per window
#define MP    64      // padded tokens (4 M-tiles of 16)
#define NWIN  64
#define NWTOT 4096    // BATCH * NWIN
#define DH    32
#define HID   512

// ---------------- WMMA fragment helpers (16x16x32 f16 -> f32) ----------------
// A (16x32 f16, row-major source, leading dim lda): lane l holds row m=l&15,
// halves 0..7 = K[8g .. 8g+7], halves 8..15 = K[16+8g .. 16+8g+7], g=l>>4.
__device__ __forceinline__ v16h frag_a(const f16* base, int lda) {
  int lane = threadIdx.x & 31;
  const f16* p0 = base + (lane & 15) * lda + 8 * (lane >> 4);
  const f16* p1 = p0 + 16;
  v16h a;
#pragma unroll
  for (int i = 0; i < 8; ++i) a[i] = p0[i];
#pragma unroll
  for (int i = 0; i < 8; ++i) a[i + 8] = p1[i];
  return a;
}
// B (32x16 f16) from B-transposed storage bt[N][K] (leading dim ldb):
// lane l holds col n=l&15, halves 0..7 = K[8g .. 8g+7], halves 8..15 = K[16+8g .. 16+8g+7].
__device__ __forceinline__ v16h frag_b(const f16* bt, int ldb) {
  int lane = threadIdx.x & 31;
  const f16* p = bt + (lane & 15) * ldb + 8 * (lane >> 4);
  v16h b;
#pragma unroll
  for (int i = 0; i < 16; ++i) b[i] = p[(i & 7) + 16 * (i >> 3)];
  return b;
}
__device__ __forceinline__ v8f wmma32(v16h a, v16h b, v8f c) {
  return __builtin_amdgcn_wmma_f32_16x16x32_f16(false, a, false, b, (short)0, c,
                                                false, false);
}
// C/D element (r, lane): row = r + 8*(lane>>4), col = lane&15.

// shift-mask region value for a rolled-image coordinate
__device__ __forceinline__ int region(int r) { return r < 49 ? 0 : (r < 53 ? 1 : 2); }

// ---------------- Kernel 1: weight transpose + f16 convert ----------------
__global__ __launch_bounds__(256) void k_wprep(
    const float* __restrict__ qkv_w, const float* __restrict__ out_w,
    const float* __restrict__ w1, const float* __restrict__ w2,
    f16* __restrict__ qkv_wt, f16* __restrict__ out_wt,
    f16* __restrict__ w1t, f16* __restrict__ w2t) {
  int i = blockIdx.x * 256 + threadIdx.x;
  if (i < 384 * 128) {                       // qkv_wt[n][k] = qkv_w[k][n]
    int n = i >> 7, k = i & 127;
    qkv_wt[i] = (f16)qkv_w[k * 384 + n];
  } else if (i < 384 * 128 + 128 * 128) {    // out_wt[n][k] = out_w[k][n]
    int j = i - 384 * 128, n = j >> 7, k = j & 127;
    out_wt[j] = (f16)out_w[k * 128 + n];
  } else if (i < 384 * 128 + 128 * 128 + 512 * 128) {  // w1t[n][k] = w1[k][n]
    int j = i - (384 * 128 + 128 * 128), n = j >> 7, k = j & 127;
    w1t[j] = (f16)w1[k * 512 + n];
  } else if (i < 384 * 128 + 128 * 128 + 512 * 128 + 128 * 512) {
    int j = i - (384 * 128 + 128 * 128 + 512 * 128);   // w2t[n][k] = w2[k][n]
    int n = j / 512, k = j % 512;
    w2t[j] = (f16)w2[k * 128 + n];
  }
}

// ---------------- Kernel 2: LN1 + roll + window partition -> f16 ----------------
__global__ __launch_bounds__(256) void k_prep(
    const float* __restrict__ x, const float* __restrict__ nw,
    const float* __restrict__ nb, f16* __restrict__ hwin) {
  int win = blockIdx.x;                      // 0..4095
  int bb = win >> 6, w2 = win & 63, wy = w2 >> 3, wx = w2 & 7;
  int wave = threadIdx.x >> 5, lane = threadIdx.x & 31;
  for (int t = wave; t < MP; t += 8) {
    f16* dst = hwin + ((size_t)win * MP + t) * DMODEL + lane * 4;
    if (t < WS) {
      int ty = t / 7, tx = t % 7;
      int sr = (wy * 7 + ty + 53) % 56, sc = (wx * 7 + tx + 53) % 56;
      const float4 v =
          *(const float4*)(x + (((size_t)bb * SDIM + sr) * SDIM + sc) * DMODEL + lane * 4);
      float s = v.x + v.y + v.z + v.w;
      float s2 = v.x * v.x + v.y * v.y + v.z * v.z + v.w * v.w;
#pragma unroll
      for (int o = 16; o; o >>= 1) {
        s += __shfl_xor(s, o, 32);
        s2 += __shfl_xor(s2, o, 32);
      }
      float mu = s * (1.f / 128.f);
      float rs = rsqrtf(s2 * (1.f / 128.f) - mu * mu + 1e-5f);
      int ch = lane * 4;
      v4h o;
      o[0] = (f16)((v.x - mu) * rs * nw[ch + 0] + nb[ch + 0]);
      o[1] = (f16)((v.y - mu) * rs * nw[ch + 1] + nb[ch + 1]);
      o[2] = (f16)((v.z - mu) * rs * nw[ch + 2] + nb[ch + 2]);
      o[3] = (f16)((v.w - mu) * rs * nw[ch + 3] + nb[ch + 3]);
      *(v4h*)dst = o;
    } else {
      v4h z = {};
      *(v4h*)dst = z;
    }
  }
}

// ---------------- Kernel 3: fused window attention ----------------
// 1 block = 1 window, 8 waves. LDS: Q/K/Vt (3*16KB) + {A-stage|S f32} (16KB)
// + attn f16 (8KB) + O f16 (16KB) = 90112 bytes.
__global__ __launch_bounds__(256) void k_attn(
    const f16* __restrict__ hwin, const f16* __restrict__ qkv_wt,
    const float* __restrict__ qkv_b, const f16* __restrict__ out_wt,
    const float* __restrict__ out_b, const float* __restrict__ rel_table,
    float* __restrict__ obuf) {
  extern __shared__ char smem[];
  f16* sQ = (f16*)smem;                  // [4][64][32]
  f16* sK = sQ + 4 * 64 * 32;            // [4][64][32]
  f16* sVt = sK + 4 * 64 * 32;           // [4][32][64]
  f16* sAw = sVt + 4 * 64 * 32;          // [64][128] staged A (dead after stage 1)
  float* sS = (float*)sAw;               // [64][64] f32 scores (reuses sAw)
  f16* sA = (f16*)(sS + 64 * 64);        // [64][64] attn probs
  f16* sO = sA + 64 * 64;                // [64][128]

  int win = blockIdx.x;
  int bb = win >> 6, w2 = win & 63, wy = w2 >> 3, wx = w2 & 7;
  int wave = threadIdx.x >> 5, lane = threadIdx.x & 31;
  int lrow = 8 * (lane >> 4), lcol = lane & 15;

  const f16* Abase = hwin + (size_t)win * MP * DMODEL;

  // Warm L2/WGP$ for the shared weight tiles (global_prefetch_b8)
  __builtin_prefetch(qkv_wt + (threadIdx.x & 191) * 256, 0, 2);
  __builtin_prefetch(out_wt + (threadIdx.x & 63) * 256, 0, 2);

  // ---- stage 0: async-DMA the 16KB A window into LDS (ASYNCcnt path) ----
  {
    unsigned ldsDst = (unsigned)(uintptr_t)(void*)sAw + threadIdx.x * 64u;
    const char* gsrc = (const char*)Abase + threadIdx.x * 64u;
    asm volatile(
        "global_load_async_to_lds_b128 %0, %1, off\n\t"
        "global_load_async_to_lds_b128 %0, %1, off offset:16\n\t"
        "global_load_async_to_lds_b128 %0, %1, off offset:32\n\t"
        "global_load_async_to_lds_b128 %0, %1, off offset:48"
        :: "v"(ldsDst), "v"(gsrc)
        : "memory");
    asm volatile("s_wait_asynccnt 0" ::: "memory");
  }
  __syncthreads();

  // ---- stage 1: QKV = A @ qkv_wt  (64x384, K=128) : 96 tiles ----
  for (int i = 0; i < 12; ++i) {
    int tid = wave * 12 + i;
    int mt = tid & 3, nt = tid >> 2;
    v8f acc = {};
#pragma unroll
    for (int kk = 0; kk < 4; ++kk)
      acc = wmma32(frag_a(sAw + mt * 16 * 128 + kk * 32, 128),
                   frag_b(qkv_wt + nt * 16 * 128 + kk * 32, 128), acc);
    int col = nt * 16 + lcol;
    float bias = qkv_b[col];
    int head = (col >> 5) & 3, dh = col & 31;
    // a 16-wide tile never crosses a 32-wide head: branch is tile-invariant
    if (col < 128) {
#pragma unroll
      for (int r = 0; r < 8; ++r) {
        int row = mt * 16 + lrow + r;
        sQ[head * 2048 + row * 32 + dh] =
            (f16)((acc[r] + bias) * 0.17677669529663687f);  // * 1/sqrt(32)
      }
    } else if (col < 256) {
#pragma unroll
      for (int r = 0; r < 8; ++r) {
        int row = mt * 16 + lrow + r;
        sK[head * 2048 + row * 32 + dh] = (f16)(acc[r] + bias);
      }
    } else {
#pragma unroll
      for (int r = 0; r < 8; ++r) {
        int row = mt * 16 + lrow + r;
        sVt[head * 2048 + dh * 64 + row] = (f16)(acc[r] + bias);  // V transposed
      }
    }
  }
  __syncthreads();   // after this, sAw region is reused as sS

  // ---- stage 2: per-head attention ----
  for (int h = 0; h < 4; ++h) {
    // scores = Q @ K^T (+ rel bias + shift mask), 16 tiles, 2 per wave
    for (int i = 0; i < 2; ++i) {
      int tid = wave * 2 + i, mt = tid >> 2, nt = tid & 3;
      v8f acc = {};
      acc = wmma32(frag_a(sQ + h * 2048 + mt * 16 * 32, 32),
                   frag_b(sK + h * 2048 + nt * 16 * 32, 32), acc);
      int ki = nt * 16 + lcol;
      int ky = ki / 7, kx = ki % 7;
      int mk = (ki < WS) ? (3 * region(wy * 7 + ky) + region(wx * 7 + kx)) : 0;
#pragma unroll
      for (int r = 0; r < 8; ++r) {
        int qi = mt * 16 + lrow + r;
        float v;
        if (ki >= WS) {
          v = -1e30f;  // padded key columns -> softmax weight 0
        } else {
          v = acc[r];
          if (qi < WS) {
            int qy = qi / 7, qx = qi % 7;
            v += rel_table[h * 169 + (qy - ky + 6) * 13 + (qx - kx + 6)];
            int mq = 3 * region(wy * 7 + qy) + region(wx * 7 + qx);
            if (mq != mk) v -= 100.f;
          }
        }
        sS[qi * 64 + ki] = v;
      }
    }
    __syncthreads();
    // softmax over rows (8 rows per wave)
    for (int i = 0; i < 8; ++i) {
      int row = wave * 8 + i;
      float v0 = sS[row * 64 + lane], v1 = sS[row * 64 + lane + 32];
      float m = fmaxf(v0, v1);
#pragma unroll
      for (int o = 16; o; o >>= 1) m = fmaxf(m, __shfl_xor(m, o, 32));
      float e0 = __expf(v0 - m), e1 = __expf(v1 - m);
      float s = e0 + e1;
#pragma unroll
      for (int o = 16; o; o >>= 1) s += __shfl_xor(s, o, 32);
      float inv = 1.f / s;
      sA[row * 64 + lane] = (f16)(e0 * inv);
      sA[row * 64 + lane + 32] = (f16)(e1 * inv);
    }
    __syncthreads();
    // O_h = attn @ V  (64x32, K=64), 8 tiles, 1 per wave
    {
      int mt = wave >> 1, nt = wave & 1;
      v8f acc = {};
#pragma unroll
      for (int kk = 0; kk < 2; ++kk)
        acc = wmma32(frag_a(sA + mt * 16 * 64 + kk * 32, 64),
                     frag_b(sVt + h * 2048 + nt * 16 * 64 + kk * 32, 64), acc);
#pragma unroll
      for (int r = 0; r < 8; ++r) {
        int row = mt * 16 + lrow + r;
        sO[row * 128 + h * 32 + nt * 16 + lcol] = (f16)acc[r];
      }
    }
    __syncthreads();
  }

  // ---- stage 3: out projection + un-partition + reverse shift ----
  for (int i = 0; i < 4; ++i) {
    int tid = wave * 4 + i, mt = tid & 3, nt = tid >> 2;
    v8f acc = {};
#pragma unroll
    for (int kk = 0; kk < 4; ++kk)
      acc = wmma32(frag_a(sO + mt * 16 * 128 + kk * 32, 128),
                   frag_b(out_wt + nt * 16 * 128 + kk * 32, 128), acc);
    int col = nt * 16 + lcol;
    float bias = out_b[col];
#pragma unroll
    for (int r = 0; r < 8; ++r) {
      int t = mt * 16 + lrow + r;
      if (t < WS) {
        int ty = t / 7, tx = t % 7;
        int dr = (wy * 7 + ty + 53) % 56, dc = (wx * 7 + tx + 53) % 56;
        obuf[(((size_t)bb * SDIM + dr) * SDIM + dc) * DMODEL + col] = acc[r] + bias;
      }
    }
  }
}

// ---------------- Kernel 4: LN2 + MLP (128 -> 512 GELU -> 128) ----------------
// 1 block = 64 rows. LDS: A f16 [64][128] (16KB) + H f16 [64][512] (64KB).
__global__ __launch_bounds__(256) void k_mlp(
    const float* __restrict__ obuf, const float* __restrict__ nw,
    const float* __restrict__ nb, const f16* __restrict__ w1t,
    const float* __restrict__ b1, const f16* __restrict__ w2t,
    const float* __restrict__ b2, float* __restrict__ out) {
  extern __shared__ char smem[];
  f16* sAm = (f16*)smem;            // [64][128]
  f16* sH = sAm + 64 * 128;         // [64][512]
  int wave = threadIdx.x >> 5, lane = threadIdx.x & 31;
  int lrow = 8 * (lane >> 4), lcol = lane & 15;
  size_t row0 = (size_t)blockIdx.x * 64;

  __builtin_prefetch(w1t + threadIdx.x * 256, 0, 2);
  __builtin_prefetch(w2t + threadIdx.x * 256, 0, 2);

  for (int i = 0; i < 8; ++i) {
    int gr = wave * 8 + i;
    const float4 v = *(const float4*)(obuf + (row0 + gr) * DMODEL + lane * 4);
    float s = v.x + v.y + v.z + v.w;
    float s2 = v.x * v.x + v.y * v.y + v.z * v.z + v.w * v.w;
#pragma unroll
    for (int o = 16; o; o >>= 1) {
      s += __shfl_xor(s, o, 32);
      s2 += __shfl_xor(s2, o, 32);
    }
    float mu = s * (1.f / 128.f);
    float rs = rsqrtf(s2 * (1.f / 128.f) - mu * mu + 1e-5f);
    int ch = lane * 4;
    v4h o;
    o[0] = (f16)((v.x - mu) * rs * nw[ch + 0] + nb[ch + 0]);
    o[1] = (f16)((v.y - mu) * rs * nw[ch + 1] + nb[ch + 1]);
    o[2] = (f16)((v.z - mu) * rs * nw[ch + 2] + nb[ch + 2]);
    o[3] = (f16)((v.w - mu) * rs * nw[ch + 3] + nb[ch + 3]);
    *(v4h*)(sAm + gr * 128 + ch) = o;
  }
  __syncthreads();

  // GEMM1: [64,128] @ [128,512] + GELU(exact)  -> sH  (128 tiles, 16/wave)
  for (int i = 0; i < 16; ++i) {
    int tid = wave * 16 + i, mt = tid & 3, nt = tid >> 2;
    v8f acc = {};
#pragma unroll
    for (int kk = 0; kk < 4; ++kk)
      acc = wmma32(frag_a(sAm + mt * 16 * 128 + kk * 32, 128),
                   frag_b(w1t + nt * 16 * 128 + kk * 32, 128), acc);
    int col = nt * 16 + lcol;
    float bias = b1[col];
#pragma unroll
    for (int r = 0; r < 8; ++r) {
      int row = mt * 16 + lrow + r;
      float v = acc[r] + bias;
      v = 0.5f * v * (1.f + erff(v * 0.70710678118654752f));
      sH[row * 512 + col] = (f16)v;
    }
  }
  __syncthreads();

  // GEMM2: [64,512] @ [512,128]  (32 tiles, 4/wave)
  for (int i = 0; i < 4; ++i) {
    int tid = wave * 4 + i, mt = tid & 3, nt = tid >> 2;
    v8f acc = {};
#pragma unroll
    for (int kk = 0; kk < 16; ++kk)
      acc = wmma32(frag_a(sH + mt * 16 * 512 + kk * 32, 512),
                   frag_b(w2t + nt * 16 * 512 + kk * 32, 512), acc);
    int col = nt * 16 + lcol;
    float bias = b2[col];
#pragma unroll
    for (int r = 0; r < 8; ++r) {
      int row = mt * 16 + lrow + r;
      out[(row0 + row) * DMODEL + col] = acc[r] + bias;
    }
  }
}

// ---------------- host launcher ----------------
extern "C" void kernel_launch(void* const* d_in, const int* in_sizes, int n_in,
                              void* d_out, int out_size, void* d_ws, size_t ws_size,
                              hipStream_t stream) {
  (void)in_sizes; (void)n_in; (void)out_size; (void)ws_size;
  const float* x     = (const float*)d_in[0];
  const float* qkv_w = (const float*)d_in[1];
  const float* qkv_b = (const float*)d_in[2];
  const float* out_w = (const float*)d_in[3];
  const float* out_b = (const float*)d_in[4];
  const float* rel   = (const float*)d_in[5];
  const float* n1w   = (const float*)d_in[6];
  const float* n1b   = (const float*)d_in[7];
  const float* n2w   = (const float*)d_in[8];
  const float* n2b   = (const float*)d_in[9];
  const float* w1    = (const float*)d_in[10];
  const float* b1    = (const float*)d_in[11];
  const float* w2    = (const float*)d_in[12];
  const float* b2    = (const float*)d_in[13];

  char* ws = (char*)d_ws;
  f16* hwin   = (f16*)ws;  ws += (size_t)NWTOT * MP * DMODEL * sizeof(f16);  // 64 MB
  f16* qkv_wt = (f16*)ws;  ws += (size_t)384 * 128 * sizeof(f16);
  f16* out_wt = (f16*)ws;  ws += (size_t)128 * 128 * sizeof(f16);
  f16* w1t    = (f16*)ws;  ws += (size_t)512 * 128 * sizeof(f16);
  f16* w2t    = (f16*)ws;  ws += (size_t)128 * 512 * sizeof(f16);
  float* obuf = (float*)ws;  // [64*56*56][128] f32, ~103 MB

  k_wprep<<<768, 256, 0, stream>>>(qkv_w, out_w, w1, w2, qkv_wt, out_wt, w1t, w2t);
  k_prep<<<NWTOT, 256, 0, stream>>>(x, n1w, n1b, hwin);
  k_attn<<<NWTOT, 256, 90112, stream>>>(hwin, qkv_wt, qkv_b, out_wt, out_b, rel, obuf);
  k_mlp<<<(BATCH * SDIM * SDIM) / 64, 256, 81920, stream>>>(obuf, n2w, n2b, w1t, b1,
                                                            w2t, b2, (float*)d_out);
}
